// babyAttention_51299089384122
// MI455X (gfx1250) — matrix operand order repaired and checked by reading, not compiled, not run on hardware
//
#include <hip/hip_runtime.h>

typedef __attribute__((ext_vector_type(16))) _Float16     v16h;
typedef __attribute__((ext_vector_type(8)))  float        v8f;
typedef __attribute__((ext_vector_type(4)))  unsigned int v4u;

#define DEVINL __device__ __forceinline__

static constexpr int Bn = 2, Tn = 4096, Cn = 512, Hn = 8, Dn = 64;

// ---------------------------------------------------------------------------
// Fragment loaders (CDNA5 WMMA 16x16x32 f16 layouts, ISA 7.12.2):
//   A 16x32: lane l -> row m=l&15, halfs[0..7]=K kb..kb+7, halfs[8..15]=K kb+16..kb+23, kb=(l>>4)*8
//   B 32x16: lane l -> col n=l&15, halfs[0..7]=K kb..kb+7, halfs[8..15]=K kb+16..kb+23, kb=(l>>4)*8
// ---------------------------------------------------------------------------
DEVINL v16h load_a_h(const _Float16* base, int ld, int lane) {
  const int m = lane & 15, kb = (lane >> 4) * 8;
  const _Float16* p = base + (size_t)m * ld + kb;
  v16h a;
  ((v4u*)&a)[0] = *(const v4u*)(p);
  ((v4u*)&a)[1] = *(const v4u*)(p + 16);
  return a;
}

DEVINL v16h load_b_h(const _Float16* base, int ld, int lane) {
  const int n = lane & 15, kb = (lane >> 4) * 8;
  const _Float16* p = base + (size_t)n * ld + kb;
  v16h b;
  ((v4u*)&b)[0] = *(const v4u*)(p);
  ((v4u*)&b)[1] = *(const v4u*)(p + 16);
  return b;
}

DEVINL v8f wmma32(v16h a, v16h b, v8f c) {
  return __builtin_amdgcn_wmma_f32_16x16x32_f16(false, a, false, b, (short)0, c,
                                                false, false);
}

// CDNA5 async copy: global -> LDS, 16B per lane, tracked by ASYNCcnt.
// Low 32 bits of a generic LDS pointer are the LDS byte address (ISA 10.2).
DEVINL void async_load_b128(void* lds_ptr, const void* gptr) {
  const unsigned lds_off = (unsigned)(size_t)lds_ptr;
  asm volatile("global_load_async_to_lds_b128 %0, %1, off"
               :: "v"(lds_off), "v"((unsigned long long)(size_t)gptr)
               : "memory");
}

// ---------------------------------------------------------------------------
// Kernel 0: bulk f32 -> f16 conversion (x and the four weight matrices).
// ---------------------------------------------------------------------------
__global__ __launch_bounds__(256) void cvt16_kernel(
    const float* __restrict__ src, _Float16* __restrict__ dst, int n) {
  const int i = (blockIdx.x * 256 + threadIdx.x) * 8;
  if (i + 8 <= n) {
    const float4 f0 = *(const float4*)(src + i);
    const float4 f1 = *(const float4*)(src + i + 4);
    union { _Float16 hh[8]; v4u u; } pk;
    pk.hh[0] = (_Float16)f0.x; pk.hh[1] = (_Float16)f0.y;
    pk.hh[2] = (_Float16)f0.z; pk.hh[3] = (_Float16)f0.w;
    pk.hh[4] = (_Float16)f1.x; pk.hh[5] = (_Float16)f1.y;
    pk.hh[6] = (_Float16)f1.z; pk.hh[7] = (_Float16)f1.w;
    *(v4u*)(dst + i) = pk.u;
  }
}

// ---------------------------------------------------------------------------
// Kernel 1: fused QKV projection.  32-row blocks (B-fragment reused by two
// A-tiles), 8 waves; wave w = head w's 64 output columns for q,k,v.
// Q,K stored [B,H,T,D] f16; V stored transposed [B,H,D,T] f16.
// ---------------------------------------------------------------------------
__global__ __launch_bounds__(256) void qkv_proj_kernel(
    const _Float16* __restrict__ x16,
    const _Float16* __restrict__ wq16, const float* __restrict__ bq,
    const _Float16* __restrict__ wk16, const float* __restrict__ bk,
    const _Float16* __restrict__ wv16, const float* __restrict__ bv,
    _Float16* __restrict__ qbuf, _Float16* __restrict__ kbuf,
    _Float16* __restrict__ vtbuf) {
  constexpr int LDX = Cn + 8;          // 520-half pitch (1040B, 16B-aligned)
  __shared__ _Float16 xs[32 * LDX];    // ~33 KB

  const int tid  = threadIdx.x;
  const int lane = tid & 31;
  const int wave = tid >> 5;           // head index
  const int row0 = blockIdx.x * 32;

  {  // stage x tile (f16 already): 8 threads/row, 64 halfs each
    const int r  = tid >> 3;
    const int c0 = (tid & 7) * 64;
    const _Float16* src = x16 + (size_t)(row0 + r) * Cn + c0;
#pragma unroll
    for (int i = 0; i < 64; i += 8)
      *(v4u*)&xs[r * LDX + c0 + i] = *(const v4u*)(src + i);
  }
  __syncthreads();

  const int b    = row0 >> 12;
  const int t0   = row0 & (Tn - 1);
  const int h    = wave;
  const int bh   = b * Hn + h;
  const int nloc = lane & 15;
  const int mb8  = (lane >> 4) * 8;

  const _Float16* Wp[3] = {wq16, wk16, wv16};
  const float*    Bp[3] = {bq, bk, bv};

#pragma unroll
  for (int proj = 0; proj < 3; ++proj) {
    const _Float16* W = Wp[proj];
    v8f acc[2][4] = {};

#pragma unroll 1
    for (int k0 = 0; k0 < Cn; k0 += 32) {
      const v16h a0 = load_a_h(&xs[k0], LDX, lane);
      const v16h a1 = load_a_h(&xs[16 * LDX + k0], LDX, lane);
      const _Float16* wrow = W + (size_t)h * 64 * Cn + k0;
      __builtin_prefetch(wrow + 64, 0, 1);
#pragma unroll
      for (int nt = 0; nt < 4; ++nt) {
        const v16h bf = load_b_h(wrow + (size_t)nt * 16 * Cn, Cn, lane);
        acc[0][nt] = wmma32(a0, bf, acc[0][nt]);
        acc[1][nt] = wmma32(a1, bf, acc[1][nt]);
      }
    }

    const float* bias = Bp[proj];
#pragma unroll
    for (int rt = 0; rt < 2; ++rt) {
      const int tr = t0 + rt * 16 + mb8;   // first of this lane's 8 rows
#pragma unroll
      for (int nt = 0; nt < 4; ++nt) {
        const int d    = nt * 16 + nloc;
        const float bb = bias[h * 64 + d];
        if (proj == 2) {
          union { _Float16 hh[8]; v4u u; } pk;
#pragma unroll
          for (int j = 0; j < 8; ++j)
            pk.hh[j] = (_Float16)(acc[rt][nt][j] + bb);
          *(v4u*)(vtbuf + ((size_t)bh * Dn + d) * Tn + tr) = pk.u;
        } else {
          _Float16* dst = (proj == 0 ? qbuf : kbuf) +
                          ((size_t)bh * Tn + tr) * Dn + d;
#pragma unroll
          for (int j = 0; j < 8; ++j)
            dst[(size_t)j * Dn] = (_Float16)(acc[rt][nt][j] + bb);
        }
      }
    }
  }
}

// ---------------------------------------------------------------------------
// Kernel 2: causal flash attention, block-cooperative K/V staging.
//   grid=(T/64, B*H), 128 thr = 4 waves; wave owns one 16-query tile.
//   K (32x64) and V^T (64x32) chunks double-buffered in LDS via
//   global_load_async_to_lds_b128 (ASYNCcnt) while WMMAs consume.
// ---------------------------------------------------------------------------
DEVINL void stage_chunk(const _Float16* kbh, const _Float16* vbh, int jc,
                        _Float16* kls, _Float16* vls, int tid) {
  constexpr int KLP = 80, VLP = 48;
#pragma unroll
  for (int i = 0; i < 2; ++i) {
    const int c = tid + i * 128;
    // K: 32 rows x 64 halfs = 256 16B-chunks
    const int kr = c >> 3, ko = (c & 7) * 8;
    async_load_b128(kls + kr * KLP + ko, kbh + (size_t)(jc + kr) * Dn + ko);
    // V^T: 64 rows x 32 halfs = 256 16B-chunks
    const int vr = c >> 2, vo = (c & 3) * 8;
    async_load_b128(vls + vr * VLP + vo, vbh + (size_t)vr * Tn + jc + vo);
  }
}

__global__ __launch_bounds__(128) void flash_attn_kernel(
    const _Float16* __restrict__ qbuf, const _Float16* __restrict__ kbuf,
    const _Float16* __restrict__ vtbuf, _Float16* __restrict__ attn) {
  constexpr int KLP = 80, VLP = 48, PST = 40;
  constexpr int KSZ = 32 * KLP, VSZ = 64 * VLP;
  __shared__ _Float16 smem[2 * KSZ + 2 * VSZ + 4 * 16 * PST];  // ~27.6 KB

  const int tid  = threadIdx.x;
  const int lane = tid & 31;
  const int wave = tid >> 5;
  const int bh   = blockIdx.y;
  const int qt   = blockIdx.x * 4 + wave;
  const int row0 = qt * 16;
  const int nloc = lane & 15;
  const int mb8  = (lane >> 4) * 8;

  const _Float16* kbh = kbuf + (size_t)bh * Tn * Dn;
  const _Float16* vbh = vtbuf + (size_t)bh * Dn * Tn;
  // double buffers selected by pointer arithmetic (no LDS-pointer arrays:
  // constant-initialized arrays of addrspacecast(LDS) break ld.lld)
  _Float16* kls0  = smem;
  _Float16* vls0  = smem + 2 * KSZ;
  _Float16* pslot = smem + 2 * KSZ + 2 * VSZ + wave * 16 * PST;

  const _Float16* Q = qbuf + ((size_t)bh * Tn + row0) * Dn;
  const v16h qa0 = load_a_h(Q, Dn, lane);       // d =  0..31
  const v16h qa1 = load_a_h(Q + 32, Dn, lane);  // d = 32..63

  v8f o0 = {}, o1 = {}, o2 = {}, o3 = {};
  float mrow[8], lrow[8];
#pragma unroll
  for (int j = 0; j < 8; ++j) { mrow[j] = -__builtin_inff(); lrow[j] = 0.0f; }

  const float scale = 0.125f;                 // 1/sqrt(64)
  const int jend = row0 + 16;                 // wave's causal key bound
  const int nch  = 2 * (blockIdx.x + 1);      // block's 32-key chunk count

  stage_chunk(kbh, vbh, 0, kls0, vls0, tid);

#pragma unroll 1
  for (int c = 0; c < nch; ++c) {
    asm volatile("s_wait_asynccnt 0" ::: "memory");
    __syncthreads();
    if (c + 1 < nch)  // prefetch next chunk into the other buffer
      stage_chunk(kbh, vbh, (c + 1) * 32, kls0 + ((c + 1) & 1) * KSZ,
                  vls0 + ((c + 1) & 1) * VSZ, tid);

    if (c * 32 < jend) {  // wave-uniform: EXEC stays all-ones for WMMA
      const int jc = c * 32;
      const _Float16* K0 = kls0 + (c & 1) * KSZ;
      const _Float16* V0 = vls0 + (c & 1) * VSZ;

      // S tiles: s0 = keys jc..jc+15, s1 = keys jc+16..jc+31
      v8f s0 = {}, s1 = {};
      s0 = wmma32(qa0, load_b_h(K0, KLP, lane), s0);
      s0 = wmma32(qa1, load_b_h(K0 + 32, KLP, lane), s0);
      s1 = wmma32(qa0, load_b_h(K0 + 16 * KLP, KLP, lane), s1);
      s1 = wmma32(qa1, load_b_h(K0 + 16 * KLP + 32, KLP, lane), s1);

      // scale + causal mask
#pragma unroll
      for (int j = 0; j < 8; ++j) {
        const int m = row0 + mb8 + j;
        float a0 = s0[j] * scale;
        float a1 = s1[j] * scale;
        if (jc + nloc > m)      a0 = -__builtin_inff();
        if (jc + 16 + nloc > m) a1 = -__builtin_inff();
        s0[j] = a0; s1[j] = a1;
      }

      // online softmax: row reductions via 16-lane butterflies
#pragma unroll
      for (int j = 0; j < 8; ++j) {
        float rm = fmaxf(s0[j], s1[j]);
#pragma unroll
        for (int off = 1; off < 16; off <<= 1)
          rm = fmaxf(rm, __shfl_xor(rm, off, 16));
        const float mnew = fmaxf(mrow[j], rm);
        const float corr = __expf(mrow[j] - mnew);
        const float p0   = __expf(s0[j] - mnew);
        const float p1   = __expf(s1[j] - mnew);
        float rs = p0 + p1;
#pragma unroll
        for (int off = 1; off < 16; off <<= 1)
          rs += __shfl_xor(rs, off, 16);
        lrow[j] = lrow[j] * corr + rs;
        mrow[j] = mnew;
        o0[j] *= corr; o1[j] *= corr; o2[j] *= corr; o3[j] *= corr;
        // stage P (C-layout -> row-major LDS tile) for A-fragment reload
        pslot[(mb8 + j) * PST + nloc]      = (_Float16)p0;
        pslot[(mb8 + j) * PST + 16 + nloc] = (_Float16)p1;
      }
      asm volatile("s_wait_dscnt 0" ::: "memory");  // intra-wave DS RAW
      const v16h pa = load_a_h(pslot, PST, lane);

      // O += P * V
      o0 = wmma32(pa, load_b_h(V0 + 0 * 16 * VLP, VLP, lane), o0);
      o1 = wmma32(pa, load_b_h(V0 + 1 * 16 * VLP, VLP, lane), o1);
      o2 = wmma32(pa, load_b_h(V0 + 2 * 16 * VLP, VLP, lane), o2);
      o3 = wmma32(pa, load_b_h(V0 + 3 * 16 * VLP, VLP, lane), o3);
    }
  }

  // epilogue: O /= l, write attn[B*T][C] at columns h*64 + d
  const int b = bh / Hn, h = bh % Hn;
#pragma unroll
  for (int j = 0; j < 8; ++j) {
    const float inv = 1.0f / lrow[j];
    const int t = row0 + mb8 + j;
    _Float16* dst = attn + ((size_t)(b * Tn + t)) * Cn + h * Dn + nloc;
    dst[0]  = (_Float16)(o0[j] * inv);
    dst[16] = (_Float16)(o1[j] * inv);
    dst[32] = (_Float16)(o2[j] * inv);
    dst[48] = (_Float16)(o3[j] * inv);
  }
}

// ---------------------------------------------------------------------------
// Kernel 3: output projection  out = attn @ wo^T + bo  (f32 result).
// ---------------------------------------------------------------------------
__global__ __launch_bounds__(256) void out_proj_kernel(
    const _Float16* __restrict__ attn, const _Float16* __restrict__ wo16,
    const float* __restrict__ bo, float* __restrict__ out) {
  constexpr int LDX = Cn + 8;
  __shared__ _Float16 xs[32 * LDX];

  const int tid  = threadIdx.x;
  const int lane = tid & 31;
  const int wave = tid >> 5;
  const int row0 = blockIdx.x * 32;

  {
    const int r  = tid >> 3;
    const int c0 = (tid & 7) * 64;
    const _Float16* src = attn + (size_t)(row0 + r) * Cn + c0;
#pragma unroll
    for (int i = 0; i < 64; i += 8)
      *(v4u*)&xs[r * LDX + c0 + i] = *(const v4u*)(src + i);
  }
  __syncthreads();

  const int n0   = wave * 64;
  const int nloc = lane & 15;
  const int mb8  = (lane >> 4) * 8;

  v8f acc[2][4] = {};
#pragma unroll 1
  for (int k0 = 0; k0 < Cn; k0 += 32) {
    const v16h a0 = load_a_h(&xs[k0], LDX, lane);
    const v16h a1 = load_a_h(&xs[16 * LDX + k0], LDX, lane);
    const _Float16* wrow = wo16 + (size_t)n0 * Cn + k0;
    __builtin_prefetch(wrow + 64, 0, 1);
#pragma unroll
    for (int nt = 0; nt < 4; ++nt) {
      const v16h bf = load_b_h(wrow + (size_t)nt * 16 * Cn, Cn, lane);
      acc[0][nt] = wmma32(a0, bf, acc[0][nt]);
      acc[1][nt] = wmma32(a1, bf, acc[1][nt]);
    }
  }

#pragma unroll
  for (int rt = 0; rt < 2; ++rt) {
#pragma unroll
    for (int nt = 0; nt < 4; ++nt) {
      const int col  = n0 + nt * 16 + nloc;
      const float bb = bo[col];
#pragma unroll
      for (int j = 0; j < 8; ++j)
        out[(size_t)(row0 + rt * 16 + mb8 + j) * Cn + col] = acc[rt][nt][j] + bb;
    }
  }
}

// ---------------------------------------------------------------------------
extern "C" void kernel_launch(void* const* d_in, const int* in_sizes, int n_in,
                              void* d_out, int out_size, void* d_ws,
                              size_t ws_size, hipStream_t stream) {
  (void)in_sizes; (void)n_in; (void)out_size; (void)ws_size;
  const float* x  = (const float*)d_in[0];
  const float* wq = (const float*)d_in[1];
  const float* bq = (const float*)d_in[2];
  const float* wk = (const float*)d_in[3];
  const float* bk = (const float*)d_in[4];
  const float* wv = (const float*)d_in[5];
  const float* bv = (const float*)d_in[6];
  const float* wo = (const float*)d_in[7];
  const float* bo = (const float*)d_in[8];
  float* out = (float*)d_out;

  // workspace layout (f16): q,k,v^T,attn (4M each) + x16 (4M) + 4 weights
  const size_t nqkv = (size_t)Bn * Hn * Tn * Dn;   // 4,194,304
  const size_t nw   = (size_t)Cn * Cn;             //   262,144
  _Float16* ws    = (_Float16*)d_ws;
  _Float16* qbuf  = ws;
  _Float16* kbuf  = ws + nqkv;
  _Float16* vtbuf = ws + 2 * nqkv;
  _Float16* abuf  = ws + 3 * nqkv;
  _Float16* x16   = ws + 4 * nqkv;
  _Float16* wq16  = ws + 5 * nqkv;
  _Float16* wk16  = wq16 + nw;
  _Float16* wv16  = wk16 + nw;
  _Float16* wo16  = wv16 + nw;

  const int nx = Bn * Tn * Cn;   // 4,194,304
  const int nwm = Cn * Cn;       //   262,144
  cvt16_kernel<<<dim3(nx / 2048),  256, 0, stream>>>(x,  x16,  nx);
  cvt16_kernel<<<dim3(nwm / 2048), 256, 0, stream>>>(wq, wq16, nwm);
  cvt16_kernel<<<dim3(nwm / 2048), 256, 0, stream>>>(wk, wk16, nwm);
  cvt16_kernel<<<dim3(nwm / 2048), 256, 0, stream>>>(wv, wv16, nwm);
  cvt16_kernel<<<dim3(nwm / 2048), 256, 0, stream>>>(wo, wo16, nwm);

  qkv_proj_kernel<<<dim3((Bn * Tn) / 32), 256, 0, stream>>>(
      x16, wq16, bq, wk16, bk, wv16, bv, qbuf, kbuf, vtbuf);
  flash_attn_kernel<<<dim3(Tn / 64, Bn * Hn), 128, 0, stream>>>(
      qbuf, kbuf, vtbuf, abuf);
  out_proj_kernel<<<dim3((Bn * Tn) / 32), 256, 0, stream>>>(abuf, wo16, bo, out);
}
